// MultiHeadAttention_29489245454730
// MI455X (gfx1250) — compile-verified, matrix operand order checked by reading, not run on hardware
//
#include <hip/hip_runtime.h>
#include <cstdint>
#include <cstddef>

// Problem constants (match reference)
static constexpr int D_   = 1024;   // d_model
static constexpr int H_   = 16;     // heads
static constexpr int DK_  = 64;     // head dim
static constexpr int B_   = 2;      // batch
static constexpr int S_   = 2048;   // seq len
static constexpr int M_   = B_ * S_;  // 4096 rows for projections
static constexpr float NEGC  = -1000000000.0f;
static constexpr float SCALE = 0.125f;  // 1/sqrt(64)

typedef __attribute__((ext_vector_type(16))) __bf16 v16bf;
typedef __attribute__((ext_vector_type(8)))  float  v8f;
typedef __attribute__((ext_vector_type(4)))  int    i32x4;

union FragAB { uint4 q[2]; v16bf v; };
union FragC  { float f[8];  v8f  v; };

#if defined(__gfx1250__) && \
    __has_builtin(__builtin_amdgcn_global_load_async_to_lds_b128) && \
    __has_builtin(__builtin_amdgcn_s_wait_asynccnt)
#define HAVE_ASYNC_LDS 1
#else
#define HAVE_ASYNC_LDS 0
#endif

__device__ __forceinline__ unsigned short f2bf(float x) {
  unsigned u = __float_as_uint(x);
  u += 0x7FFFu + ((u >> 16) & 1u);   // round-to-nearest-even
  return (unsigned short)(u >> 16);
}

__device__ __forceinline__ v8f wmma_bf16(v16bf a, v16bf b, v8f c) {
  // D = A(16x32 bf16) * B(32x16 bf16) + C(16x16 f32)
  return __builtin_amdgcn_wmma_f32_16x16x32_bf16(false, a, false, b,
                                                 (short)0, c, false, false);
}

// 16-byte global -> LDS copy: async (ASYNCcnt) if available, else load+store.
__device__ __forceinline__ void copy16_g2l(const unsigned short* g, unsigned short* l) {
#if HAVE_ASYNC_LDS
  typedef i32x4 __attribute__((address_space(1)))* gvec_p;
  typedef i32x4 __attribute__((address_space(3)))* lvec_p;
  __builtin_amdgcn_global_load_async_to_lds_b128(
      (gvec_p)const_cast<unsigned short*>(g), (lvec_p)l, 0, 0);
#else
  *(uint4*)l = *(const uint4*)g;
#endif
}

__device__ __forceinline__ void wait_async_done() {
#if HAVE_ASYNC_LDS
  __builtin_amdgcn_s_wait_asynccnt(0);
#endif
}

// ---------------------------------------------------------------------------
// fp32 -> bf16 conversion, 4 elements per thread, exact grid
// ---------------------------------------------------------------------------
__global__ __launch_bounds__(256)
void cvt_f32_bf16(const float* __restrict__ src, unsigned short* __restrict__ dst, int n) {
  int i = (blockIdx.x * 256 + threadIdx.x) * 4;
  if (i + 3 < n) {
    float4 f = *(const float4*)(src + i);
    ushort4 o;
    o.x = f2bf(f.x); o.y = f2bf(f.y); o.z = f2bf(f.z); o.w = f2bf(f.w);
    *(ushort4*)(dst + i) = o;
  }
}

// ---------------------------------------------------------------------------
// Y[m][n] = sum_k X[m][k] * W[n][k]   (torch Linear: x @ W.T)
// M=4096, N=K=1024. Block tile 128x128, 8 waves, wave tile 64x32.
// mode 0: bf16 out, head-split row-major  dst[((b*H+h)*S+s)*64+d]
// mode 1: bf16 out, head-split transposed dst[((b*H+h)*64+d)*S+s]   (for V)
// mode 2: f32 out + bias, row-major        dst[m*D+n]
// ---------------------------------------------------------------------------
__global__ __launch_bounds__(256)
void proj_gemm(const unsigned short* __restrict__ X,
               const unsigned short* __restrict__ W,
               void* __restrict__ dstv,
               const float* __restrict__ bias,
               int mode)
{
  const int lane = threadIdx.x & 31;
  const int wave = threadIdx.x >> 5;
  const int half = lane >> 4;       // which K-half this lane holds
  const int l15  = lane & 15;
  const int mBase = blockIdx.y * 128 + (wave >> 2) * 64;
  const int nBase = blockIdx.x * 128 + (wave & 3) * 32;

  FragC acc[4][2];
  #pragma unroll
  for (int mt = 0; mt < 4; ++mt)
    #pragma unroll
    for (int nt = 0; nt < 2; ++nt)
      #pragma unroll
      for (int r = 0; r < 8; ++r) acc[mt][nt].f[r] = 0.0f;

  for (int k = 0; k < D_; k += 32) {
    FragAB a[4], b[2];
    #pragma unroll
    for (int mt = 0; mt < 4; ++mt) {
      const uint4* p = (const uint4*)(X + (size_t)(mBase + mt*16 + l15) * D_ + k + half*8);
      a[mt].q[0] = p[0];   // K = kbase+0..7
      a[mt].q[1] = p[2];   // K = kbase+16..23
    }
    #pragma unroll
    for (int nt = 0; nt < 2; ++nt) {
      const uint4* p = (const uint4*)(W + (size_t)(nBase + nt*16 + l15) * D_ + k + half*8);
      b[nt].q[0] = p[0];
      b[nt].q[1] = p[2];
    }
    if (k + 32 < D_) {
      __builtin_prefetch(X + (size_t)(mBase + l15) * D_ + k + 32, 0, 0);
      __builtin_prefetch(W + (size_t)(nBase + l15) * D_ + k + 32, 0, 0);
    }
    #pragma unroll
    for (int mt = 0; mt < 4; ++mt)
      #pragma unroll
      for (int nt = 0; nt < 2; ++nt)
        acc[mt][nt].v = wmma_bf16(a[mt].v, b[nt].v, acc[mt][nt].v);
  }

  #pragma unroll
  for (int mt = 0; mt < 4; ++mt) {
    #pragma unroll
    for (int nt = 0; nt < 2; ++nt) {
      #pragma unroll
      for (int r = 0; r < 8; ++r) {
        const int m = mBase + mt*16 + half*8 + r;   // C/D layout: row
        const int n = nBase + nt*16 + l15;          // C/D layout: col
        const float val = acc[mt][nt].f[r];
        if (mode == 2) {
          ((float*)dstv)[(size_t)m * D_ + n] = val + bias[n];
        } else {
          const int bb = m >> 11, s = m & (S_ - 1);
          const int hh = n >> 6,  d = n & (DK_ - 1);
          unsigned short* dst = (unsigned short*)dstv;
          if (mode == 0)
            dst[(((size_t)(bb*H_ + hh)) * S_ + s) * DK_ + d] = f2bf(val);
          else
            dst[(((size_t)(bb*H_ + hh)) * DK_ + d) * S_ + s] = f2bf(val);
        }
      }
    }
  }
}

// ---------------------------------------------------------------------------
// Flash attention, LDS-staged K/V shared by all 8 waves of the workgroup.
// One wave = 16 query rows of one (b,h); WG covers 128 query rows.
// Qh,Kh: [B*H][S][64] bf16.  Vt: [B*H][64][S] bf16.  mask: [B][S][S] i32.
// O: [B][S][H*64] bf16.
// Per 32-key step: double-buffered K tile (32x64) + V tile (64x32) staged
// via async-to-LDS (ASYNCcnt) with a one-barrier software pipeline.
// ---------------------------------------------------------------------------
__global__ __launch_bounds__(256)
void attn_flash(const unsigned short* __restrict__ Qh,
                const unsigned short* __restrict__ Kh,
                const unsigned short* __restrict__ Vt,
                const int* __restrict__ mask,
                unsigned short* __restrict__ O)
{
  __shared__ unsigned short ktile[2][32][64];   // [buf][key][dk]      4KB each
  __shared__ unsigned short vtile[2][64][32];   // [buf][dk][key]      4KB each
  __shared__ unsigned short plds[8][16][32];    // per-wave P tile     8KB total

  const int tid  = threadIdx.x;
  const int lane = tid & 31;
  const int wave = tid >> 5;
  const int half = lane >> 4;
  const int l15  = lane & 15;

  const int bh = blockIdx.y;
  const int b  = bh >> 4;
  const int h  = bh & 15;
  const int qBase = blockIdx.x * 128 + wave * 16;

  const unsigned short* Qp = Qh + ((size_t)bh * S_ + qBase) * DK_;
  const unsigned short* Kp = Kh + (size_t)bh * S_ * DK_;
  const unsigned short* Vp = Vt + (size_t)bh * DK_ * S_;

  // Staging assignment: 256 threads, 16B each.
  // K tile: 32 rows x 64 bf16 = 256 chunks -> row=tid>>3, chunk=tid&7
  // V tile: 64 rows x 32 bf16 = 256 chunks -> row=tid>>2, chunk=tid&3
  const int krow = tid >> 3, kch = (tid & 7) * 8;
  const int vrow = tid >> 2, vch = (tid & 3) * 8;

  // Q fragments (16 x 64 = two 16x32 A fragments), loaded once
  FragAB qf[2];
  #pragma unroll
  for (int kk = 0; kk < 2; ++kk) {
    const uint4* p = (const uint4*)(Qp + (size_t)l15 * DK_ + kk*32 + half*8);
    qf[kk].q[0] = p[0];
    qf[kk].q[1] = p[2];
  }

  FragC o[4];
  #pragma unroll
  for (int nt = 0; nt < 4; ++nt)
    #pragma unroll
    for (int r = 0; r < 8; ++r) o[nt].f[r] = 0.0f;

  float mrun[8], lrun[8];
  #pragma unroll
  for (int r = 0; r < 8; ++r) { mrun[r] = -1e30f; lrun[r] = 0.0f; }

  // Prologue: stage tile for j=0 into buffer 0
  copy16_g2l(Kp + (size_t)krow * DK_ + kch, &ktile[0][krow][kch]);
  copy16_g2l(Vp + (size_t)vrow * S_  + vch, &vtile[0][vrow][vch]);

  for (int j = 0; j < S_; j += 32) {
    const int cur = (j >> 5) & 1;

    wait_async_done();     // this wave's copies into buf[cur] are complete
    __syncthreads();       // => whole tile[cur] ready; everyone done with buf[cur^1]

    if (j + 32 < S_) {     // stage next tile into the other buffer
      const int nxt = cur ^ 1;
      copy16_g2l(Kp + (size_t)(j + 32 + krow) * DK_ + kch, &ktile[nxt][krow][kch]);
      copy16_g2l(Vp + (size_t)vrow * S_ + (j + 32) + vch,  &vtile[nxt][vrow][vch]);
    }

    // ---- scores: two 16x16 tiles covering keys [j, j+32) ----
    FragC s0, s1;
    #pragma unroll
    for (int r = 0; r < 8; ++r) { s0.f[r] = 0.0f; s1.f[r] = 0.0f; }
    #pragma unroll
    for (int kk = 0; kk < 2; ++kk) {             // dk contraction in two steps
      FragAB kf0, kf1;
      const uint4* p0 = (const uint4*)&ktile[cur][l15][kk*32 + half*8];
      kf0.q[0] = p0[0]; kf0.q[1] = p0[2];
      const uint4* p1 = (const uint4*)&ktile[cur][16 + l15][kk*32 + half*8];
      kf1.q[0] = p1[0]; kf1.q[1] = p1[2];
      s0.v = wmma_bf16(qf[kk].v, kf0.v, s0.v);
      s1.v = wmma_bf16(qf[kk].v, kf1.v, s1.v);
    }

    // ---- mask + scale + online softmax (rows live in 16-lane halves) ----
    #pragma unroll
    for (int r = 0; r < 8; ++r) {
      const int qrow = qBase + half*8 + r;
      const int* Mrow = mask + ((size_t)b * S_ + qrow) * S_;
      float v0 = s0.f[r] * SCALE;
      float v1 = s1.f[r] * SCALE;
      if (Mrow[j + l15]      == 0) v0 = NEGC;
      if (Mrow[j + 16 + l15] == 0) v1 = NEGC;
      float mx = fmaxf(v0, v1);
      #pragma unroll
      for (int off = 1; off < 16; off <<= 1)
        mx = fmaxf(mx, __shfl_xor(mx, off, 32));
      const float mn    = fmaxf(mrun[r], mx);
      const float alpha = __expf(mrun[r] - mn);
      const float p0    = __expf(v0 - mn);
      const float p1    = __expf(v1 - mn);
      mrun[r] = mn;
      float rs = p0 + p1;
      #pragma unroll
      for (int off = 1; off < 16; off <<= 1)
        rs += __shfl_xor(rs, off, 32);
      lrun[r] = lrun[r] * alpha + rs;
      #pragma unroll
      for (int nt = 0; nt < 4; ++nt) o[nt].f[r] *= alpha;   // rescale O rows
      // C/D layout -> LDS (row-major 16x32 bf16)
      plds[wave][half*8 + r][l15]      = f2bf(p0);
      plds[wave][half*8 + r][16 + l15] = f2bf(p1);
    }

    // ---- reload P as an A-matrix fragment (same-wave LDS ops are in-order) ----
    FragAB pf;
    const uint4* pp = (const uint4*)&plds[wave][l15][half*8];
    pf.q[0] = pp[0];
    pf.q[1] = pp[2];

    // ---- O += P * V  (V tile transposed in LDS: contiguous along keys) ----
    #pragma unroll
    for (int nt = 0; nt < 4; ++nt) {
      FragAB vf;
      const uint4* vp = (const uint4*)&vtile[cur][nt*16 + l15][half*8];
      vf.q[0] = vp[0];
      vf.q[1] = vp[2];
      o[nt].v = wmma_bf16(pf.v, vf.v, o[nt].v);
    }
  }

  // ---- normalize and store: O[b][s][h*64+d] bf16 ----
  #pragma unroll
  for (int nt = 0; nt < 4; ++nt) {
    #pragma unroll
    for (int r = 0; r < 8; ++r) {
      const int s = qBase + half*8 + r;
      const float inv = 1.0f / lrun[r];
      O[((size_t)b * S_ + s) * D_ + h * DK_ + nt*16 + l15] = f2bf(o[nt].f[r] * inv);
    }
  }
}

// ---------------------------------------------------------------------------
extern "C" void kernel_launch(void* const* d_in, const int* in_sizes, int n_in,
                              void* d_out, int out_size, void* d_ws, size_t ws_size,
                              hipStream_t stream)
{
  (void)in_sizes; (void)n_in; (void)out_size; (void)ws_size;

  const float* q   = (const float*)d_in[0];
  const float* k   = (const float*)d_in[1];
  const float* v   = (const float*)d_in[2];
  const int*   msk = (const int*)d_in[3];
  const float* Wq  = (const float*)d_in[4];
  const float* Wk  = (const float*)d_in[5];
  const float* Wv  = (const float*)d_in[6];
  const float* Wo  = (const float*)d_in[7];
  const float* bo  = (const float*)d_in[8];
  float* out = (float*)d_out;

  const size_t NX = (size_t)M_ * D_;  // 4,194,304 elems
  const size_t NW = (size_t)D_ * D_;  // 1,048,576 elems

  unsigned short* ws  = (unsigned short*)d_ws;
  unsigned short* qbf = ws;            // [M, D] bf16
  unsigned short* kbf = qbf + NX;
  unsigned short* vbf = kbf + NX;
  unsigned short* wqb = vbf + NX;      // [D, D] bf16 each
  unsigned short* wkb = wqb + NW;
  unsigned short* wvb = wkb + NW;
  unsigned short* wob = wvb + NW;
  unsigned short* QhP = wob + NW;      // [B*H][S][64]
  unsigned short* KhP = QhP + NX;      // [B*H][S][64]
  unsigned short* VtP = KhP + NX;      // [B*H][64][S]
  unsigned short* Obf = qbf;           // reuse: q bf16 dead after Q projection
  // total ws usage: (6*NX + 4*NW) * 2 bytes = 56 MiB

  // 1) fp32 -> bf16
  cvt_f32_bf16<<<NX/1024, 256, 0, stream>>>(q,  qbf, (int)NX);
  cvt_f32_bf16<<<NX/1024, 256, 0, stream>>>(k,  kbf, (int)NX);
  cvt_f32_bf16<<<NX/1024, 256, 0, stream>>>(v,  vbf, (int)NX);
  cvt_f32_bf16<<<NW/1024, 256, 0, stream>>>(Wq, wqb, (int)NW);
  cvt_f32_bf16<<<NW/1024, 256, 0, stream>>>(Wk, wkb, (int)NW);
  cvt_f32_bf16<<<NW/1024, 256, 0, stream>>>(Wv, wvb, (int)NW);
  cvt_f32_bf16<<<NW/1024, 256, 0, stream>>>(Wo, wob, (int)NW);

  // 2) Q/K/V projections (WMMA GEMMs)
  dim3 gp(D_/128, M_/128);   // (8, 32)
  proj_gemm<<<gp, 256, 0, stream>>>(qbf, wqb, QhP, nullptr, 0);
  proj_gemm<<<gp, 256, 0, stream>>>(kbf, wkb, KhP, nullptr, 0);
  proj_gemm<<<gp, 256, 0, stream>>>(vbf, wvb, VtP, nullptr, 1);

  // 3) flash attention (LDS-staged, async double-buffered K/V)
  attn_flash<<<dim3(S_/128, B_*H_), 256, 0, stream>>>(QhP, KhP, VtP, msk, Obf);

  // 4) output projection + bias (f32 out)
  proj_gemm<<<gp, 256, 0, stream>>>(Obf, wob, out, bo, 2);
}
